// MFT_89833535963644
// MI455X (gfx1250) — matrix operand order repaired and checked by reading, not compiled, not run on hardware
//
#include <hip/hip_runtime.h>
#include <hip/hip_bf16.h>
#include <math.h>
#include <stdint.h>

// ---------------- problem constants ----------------
#define NB   8
#define NM   1024
#define ND   768
#define NL   12
#define NH   12
#define NHD  64
#define NMLP 3072
#define NV   101
#define NDX  512
#define NFREQ 256

typedef __attribute__((ext_vector_type(16))) __bf16 v16bf;
typedef __attribute__((ext_vector_type(8)))  float  v8f;
typedef __attribute__((ext_vector_type(4)))  __bf16 bf16x4;
typedef unsigned int tdm_v4u __attribute__((ext_vector_type(4)));
typedef int          tdm_v8i __attribute__((ext_vector_type(8)));
typedef int          tdm_v4i __attribute__((ext_vector_type(4)));

// ---------------- helpers ----------------
__device__ __forceinline__ float geluf(float x) {
  float x3 = x * x * x;
  return 0.5f * x * (1.0f + tanhf(0.7978845608028654f * (x + 0.044715f * x3)));
}
__device__ __forceinline__ float siluf(float x) {
  return x * (1.0f / (1.0f + __expf(-x)));
}

// ---------------- Tensor Data Mover: 2D fp32 tile -> LDS ----------------
// Builds D# per CDNA5 ISA 8.3/8.4: group0 {count=1, lds_addr, global_addr,
// type=2}; group1 {data_size=4B, tensor dims, tile dims, dim0 stride}.
__device__ __forceinline__ void tdm_load_2d(uint32_t lds_byte_off,
                                            const float* gptr,
                                            uint32_t tile_w, uint32_t tile_h,
                                            uint32_t stride_elems) {
  uint64_t ga = (uint64_t)(uintptr_t)gptr;
  tdm_v4u g0;
  g0[0] = 1u;                                   // count=1 (valid), user mode
  g0[1] = lds_byte_off;                         // lds_addr (flat low 32 = LDS offset)
  g0[2] = (uint32_t)(ga & 0xffffffffu);         // global_addr[31:0]
  g0[3] = (uint32_t)((ga >> 32) & 0x01ffffffu) | (2u << 30); // [56:32] | type=2
  tdm_v8i g1;
  g1[0] = (int)(2u << 16);                      // wg_mask=0, data_size=2 (4B)
  g1[1] = (int)(tile_w << 16);                  // tensor_dim0[15:0] @ bits63:48
  g1[2] = (int)((tile_w >> 16) | (tile_h << 16)); // dim0 hi | dim1 lo
  g1[3] = (int)((tile_h >> 16) | (tile_w << 16)); // dim1 hi | tile_dim0
  g1[4] = (int)tile_h;                          // tile_dim1 (tile_dim2=0)
  g1[5] = (int)stride_elems;                    // tensor_dim0_stride[31:0]
  g1[6] = 0;
  g1[7] = 0;
  tdm_v4i gz = {0, 0, 0, 0};
#if defined(__clang_major__) && (__clang_major__ >= 23)
  tdm_v8i gz8 = {0, 0, 0, 0, 0, 0, 0, 0};
  __builtin_amdgcn_tensor_load_to_lds(g0, g1, gz, gz, gz8, 0);
#else
  __builtin_amdgcn_tensor_load_to_lds(g0, g1, gz, gz, 0);
#endif
}

// =====================================================================
// Fast bf16 WMMA GEMM (register-pipelined staging).
// REQUIRES: M % 128 == 0, N % 128 == 0, K % 32 == 0.
// block = 256 threads (8 waves), tile 128x128, K step 32.
// =====================================================================
__global__ void gemm_wmma_fast(const float* __restrict__ A,
                               const float* __restrict__ W,
                               const float* __restrict__ bias,
                               float* __restrict__ C,
                               int M, int N, int K, int act) {
  __shared__ __bf16 As[128 * 32];   // 8 KB
  __shared__ __bf16 Ws[32 * 128];   // 8 KB

  const int tid  = threadIdx.x;
  const int lane = tid & 31;
  const int wave = tid >> 5;
  const int waveR = wave >> 2;
  const int waveC = wave & 3;
  const int rowBase = blockIdx.y * 128;
  const int colBase = blockIdx.x * 128;
  const int half = lane >> 4;
  const int mrow = lane & 15;

  // per-thread staging coordinates
  const int arA[4] = { (0*256+tid) >> 3, (1*256+tid) >> 3,
                       (2*256+tid) >> 3, (3*256+tid) >> 3 };
  const int acA[4] = { ((0*256+tid) & 7) * 4, ((1*256+tid) & 7) * 4,
                       ((2*256+tid) & 7) * 4, ((3*256+tid) & 7) * 4 };
  const int arW[4] = { (0*256+tid) >> 5, (1*256+tid) >> 5,
                       (2*256+tid) >> 5, (3*256+tid) >> 5 };
  const int acW[4] = { ((0*256+tid) & 31) * 4, ((1*256+tid) & 31) * 4,
                       ((2*256+tid) & 31) * 4, ((3*256+tid) & 31) * 4 };

  v8f acc[4][2];
#pragma unroll
  for (int mi = 0; mi < 4; ++mi)
#pragma unroll
    for (int ni = 0; ni < 2; ++ni)
#pragma unroll
      for (int r = 0; r < 8; ++r) acc[mi][ni][r] = 0.0f;

  float4 ra[4], rw[4];
#pragma unroll
  for (int it = 0; it < 4; ++it) {
    ra[it] = *(const float4*)&A[(size_t)(rowBase + arA[it]) * K + acA[it]];
    rw[it] = *(const float4*)&W[(size_t)arW[it] * N + colBase + acW[it]];
  }

  for (int k0 = 0; k0 < K; k0 += 32) {
    // commit staged registers to LDS as bf16
#pragma unroll
    for (int it = 0; it < 4; ++it) {
      bf16x4 pa, pw;
      pa[0] = (__bf16)ra[it].x; pa[1] = (__bf16)ra[it].y;
      pa[2] = (__bf16)ra[it].z; pa[3] = (__bf16)ra[it].w;
      *(bf16x4*)&As[arA[it] * 32 + acA[it]] = pa;
      pw[0] = (__bf16)rw[it].x; pw[1] = (__bf16)rw[it].y;
      pw[2] = (__bf16)rw[it].z; pw[3] = (__bf16)rw[it].w;
      *(bf16x4*)&Ws[arW[it] * 128 + acW[it]] = pw;
    }
    __syncthreads();

    // prefetch next K tile into registers (overlaps the WMMAs below)
    if (k0 + 32 < K) {
#pragma unroll
      for (int it = 0; it < 4; ++it) {
        ra[it] = *(const float4*)&A[(size_t)(rowBase + arA[it]) * K + k0 + 32 + acA[it]];
        rw[it] = *(const float4*)&W[(size_t)(k0 + 32 + arW[it]) * N + colBase + acW[it]];
      }
    }

    v16bf afrag[4], bfrag[2];
#pragma unroll
    for (int mi = 0; mi < 4; ++mi) {
      int rr = waveR * 64 + mi * 16 + mrow;
#pragma unroll
      for (int i = 0; i < 16; ++i) {
        int kk = (i >> 3) * 16 + half * 8 + (i & 7); // ISA A 16x32 layout
        afrag[mi][i] = As[rr * 32 + kk];
      }
    }
#pragma unroll
    for (int ni = 0; ni < 2; ++ni) {
      int cc = waveC * 32 + ni * 16;
#pragma unroll
      for (int i = 0; i < 16; ++i)
        bfrag[ni][i] = Ws[((i >> 3) * 16 + half * 8 + (i & 7)) * 128 + cc + mrow];
    }
#pragma unroll
    for (int mi = 0; mi < 4; ++mi)
#pragma unroll
      for (int ni = 0; ni < 2; ++ni)
        acc[mi][ni] = __builtin_amdgcn_wmma_f32_16x16x32_bf16(
            false, afrag[mi], false, bfrag[ni], (short)0, acc[mi][ni],
            false, false);
    __syncthreads();
  }

#pragma unroll
  for (int mi = 0; mi < 4; ++mi)
#pragma unroll
    for (int ni = 0; ni < 2; ++ni) {
      int colT = colBase + waveC * 32 + ni * 16 + mrow;
      float bv = bias ? bias[colT] : 0.0f;
#pragma unroll
      for (int r = 0; r < 8; ++r) {
        int row = rowBase + waveR * 64 + mi * 16 + r + 8 * half;
        float v = acc[mi][ni][r] + bv;
        if (act == 1) v = geluf(v);
        C[(size_t)row * N + colT] = v;
      }
    }
}

// =====================================================================
// TDM-staged WMMA GEMM: fp32 tiles DMA'd to LDS by the Tensor Data Mover,
// bf16 conversion at fragment read. Same tiling as gemm_wmma_fast.
// REQUIRES: M % 128 == 0, N % 128 == 0, K % 32 == 0.
// =====================================================================
__global__ void gemm_wmma_tdm(const float* __restrict__ A,
                              const float* __restrict__ W,
                              const float* __restrict__ bias,
                              float* __restrict__ C,
                              int M, int N, int K, int act) {
  __shared__ float Asf[128 * 32];   // 16 KB
  __shared__ float Wsf[32 * 128];   // 16 KB

  const int tid  = threadIdx.x;
  const int lane = tid & 31;
  const int wave = tid >> 5;
  const int waveR = wave >> 2;
  const int waveC = wave & 3;
  const int rowBase = blockIdx.y * 128;
  const int colBase = blockIdx.x * 128;
  const int half = lane >> 4;
  const int mrow = lane & 15;

  const uint32_t ldsA = (uint32_t)(uintptr_t)&Asf[0];
  const uint32_t ldsW = (uint32_t)(uintptr_t)&Wsf[0];

  v8f acc[4][2];
#pragma unroll
  for (int mi = 0; mi < 4; ++mi)
#pragma unroll
    for (int ni = 0; ni < 2; ++ni)
#pragma unroll
      for (int r = 0; r < 8; ++r) acc[mi][ni][r] = 0.0f;

  for (int k0 = 0; k0 < K; k0 += 32) {
    if (wave == 0) {
      // A tile: 128 rows x 32 cols, row pitch K ; W tile: 32 rows x 128, pitch N
      tdm_load_2d(ldsA, &A[(size_t)rowBase * K + k0], 32u, 128u, (uint32_t)K);
      tdm_load_2d(ldsW, &W[(size_t)k0 * N + colBase], 128u, 32u, (uint32_t)N);
      __builtin_amdgcn_s_wait_tensorcnt(0);
    }
    __syncthreads();

    v16bf afrag[4], bfrag[2];
#pragma unroll
    for (int mi = 0; mi < 4; ++mi) {
      int rr = waveR * 64 + mi * 16 + mrow;
#pragma unroll
      for (int i = 0; i < 16; ++i) {
        int kk = (i >> 3) * 16 + half * 8 + (i & 7);
        afrag[mi][i] = (__bf16)Asf[rr * 32 + kk];
      }
    }
#pragma unroll
    for (int ni = 0; ni < 2; ++ni) {
      int cc = waveC * 32 + ni * 16;
#pragma unroll
      for (int i = 0; i < 16; ++i)
        bfrag[ni][i] = (__bf16)Wsf[((i >> 3) * 16 + half * 8 + (i & 7)) * 128 + cc + mrow];
    }
#pragma unroll
    for (int mi = 0; mi < 4; ++mi)
#pragma unroll
      for (int ni = 0; ni < 2; ++ni)
        acc[mi][ni] = __builtin_amdgcn_wmma_f32_16x16x32_bf16(
            false, afrag[mi], false, bfrag[ni], (short)0, acc[mi][ni],
            false, false);
    __syncthreads();
  }

#pragma unroll
  for (int mi = 0; mi < 4; ++mi)
#pragma unroll
    for (int ni = 0; ni < 2; ++ni) {
      int colT = colBase + waveC * 32 + ni * 16 + mrow;
      float bv = bias ? bias[colT] : 0.0f;
#pragma unroll
      for (int r = 0; r < 8; ++r) {
        int row = rowBase + waveR * 64 + mi * 16 + r + 8 * half;
        float v = acc[mi][ni][r] + bv;
        if (act == 1) v = geluf(v);
        C[(size_t)row * N + colT] = v;
      }
    }
}

// =====================================================================
// N-guarded variant (for the V=101 logits head).
// =====================================================================
__global__ void gemm_wmma_nck(const float* __restrict__ A,
                              const float* __restrict__ W,
                              const float* __restrict__ bias,
                              float* __restrict__ C,
                              int M, int N, int K, int act) {
  __shared__ __bf16 As[128 * 32];
  __shared__ __bf16 Ws[32 * 128];

  const int tid  = threadIdx.x;
  const int lane = tid & 31;
  const int wave = tid >> 5;
  const int waveR = wave >> 2;
  const int waveC = wave & 3;
  const int rowBase = blockIdx.y * 128;
  const int colBase = blockIdx.x * 128;
  const int half = lane >> 4;
  const int mrow = lane & 15;

  v8f acc[4][2];
#pragma unroll
  for (int mi = 0; mi < 4; ++mi)
#pragma unroll
    for (int ni = 0; ni < 2; ++ni)
#pragma unroll
      for (int r = 0; r < 8; ++r) acc[mi][ni][r] = 0.0f;

  for (int k0 = 0; k0 < K; k0 += 32) {
#pragma unroll
    for (int it = 0; it < 4; ++it) {
      int idx4 = it * 256 + tid;
      int r = idx4 >> 3, c4 = idx4 & 7;
      float4 f = *(const float4*)&A[(size_t)(rowBase + r) * K + k0 + c4 * 4];
      bf16x4 pk;
      pk[0] = (__bf16)f.x; pk[1] = (__bf16)f.y;
      pk[2] = (__bf16)f.z; pk[3] = (__bf16)f.w;
      *(bf16x4*)&As[r * 32 + c4 * 4] = pk;
    }
#pragma unroll
    for (int idx = tid; idx < 32 * 128; idx += 256) {
      int r = idx >> 7, c = idx & 127;
      int gc = colBase + c;
      int gcc = gc < N ? gc : N - 1;
      float v = W[(size_t)(k0 + r) * N + gcc];
      Ws[idx] = (__bf16)(gc < N ? v : 0.0f);
    }
    __syncthreads();

    v16bf afrag[4], bfrag[2];
#pragma unroll
    for (int mi = 0; mi < 4; ++mi) {
      int rr = waveR * 64 + mi * 16 + mrow;
#pragma unroll
      for (int i = 0; i < 16; ++i) {
        int kk = (i >> 3) * 16 + half * 8 + (i & 7);
        afrag[mi][i] = As[rr * 32 + kk];
      }
    }
#pragma unroll
    for (int ni = 0; ni < 2; ++ni) {
      int cc = waveC * 32 + ni * 16;
#pragma unroll
      for (int i = 0; i < 16; ++i)
        bfrag[ni][i] = Ws[((i >> 3) * 16 + half * 8 + (i & 7)) * 128 + cc + mrow];
    }
#pragma unroll
    for (int mi = 0; mi < 4; ++mi)
#pragma unroll
      for (int ni = 0; ni < 2; ++ni)
        acc[mi][ni] = __builtin_amdgcn_wmma_f32_16x16x32_bf16(
            false, afrag[mi], false, bfrag[ni], (short)0, acc[mi][ni],
            false, false);
    __syncthreads();
  }

#pragma unroll
  for (int mi = 0; mi < 4; ++mi)
#pragma unroll
    for (int ni = 0; ni < 2; ++ni) {
      int colT = colBase + waveC * 32 + ni * 16 + mrow;
      if (colT < N) {
        float bv = bias ? bias[colT] : 0.0f;
#pragma unroll
        for (int r = 0; r < 8; ++r) {
          int row = rowBase + waveR * 64 + mi * 16 + r + 8 * half;
          float v = acc[mi][ni][r] + bv;
          if (act == 1) v = geluf(v);
          C[(size_t)row * N + colT] = v;
        }
      }
    }
}

// =====================================================================
// WMMA flash attention. q,v: (B*H, M, HD); k TRANSPOSED: (B*H, HD, M).
// One wave per 16-query tile; block = 4 waves; grid = B*H*(M/64).
// =====================================================================
__global__ void attn_wmma(const float* __restrict__ q,
                          const float* __restrict__ kT,
                          const float* __restrict__ v,
                          const unsigned char* __restrict__ mask,
                          float* __restrict__ o) {
  __shared__ float Pw[4][16][32];

  const int lane = threadIdx.x & 31;
  const int wave = threadIdx.x >> 5;
  const int tiles = NM / 64;
  const int bh = blockIdx.x / tiles;
  const int b = bh / NH, h = bh % NH;
  const int qbase = (blockIdx.x % tiles) * 64 + wave * 16;
  const int half = lane >> 4, mrow = lane & 15;
  const float scale = 0.125f; // 1/sqrt(64)

  const float* qp = q + (size_t)bh * NM * NHD;
  const float* kp = kT + (size_t)bh * NHD * NM;   // (HD, M)
  const float* vp = v + (size_t)bh * NM * NHD;

  v16bf aq[2];
#pragma unroll
  for (int ch = 0; ch < 2; ++ch)
#pragma unroll
    for (int i = 0; i < 16; ++i) {
      int kk = ch * 32 + (i >> 3) * 16 + half * 8 + (i & 7);
      aq[ch][i] = (__bf16)(qp[(size_t)(qbase + mrow) * NHD + kk] * scale);
    }

  float rmax[8], rsum[8];
#pragma unroll
  for (int r = 0; r < 8; ++r) { rmax[r] = -1e30f; rsum[r] = 0.0f; }
  v8f accO[4];
#pragma unroll
  for (int t = 0; t < 4; ++t)
#pragma unroll
    for (int r = 0; r < 8; ++r) accO[t][r] = 0.0f;

  for (int kt = 0; kt < NM / 32; ++kt) {
    v8f s0, s1;
#pragma unroll
    for (int r = 0; r < 8; ++r) { s0[r] = 0.0f; s1[r] = 0.0f; }
    float cb0, cb1;
    {
      int keyb = kt * 32;
      v16bf bk0, bk1;
#pragma unroll
      for (int i = 0; i < 16; ++i) {
        bk0[i] = (__bf16)kp[(size_t)((i >> 3) * 16 + half * 8 + (i & 7)) * NM + keyb + mrow];
        bk1[i] = (__bf16)kp[(size_t)(32 + (i >> 3) * 16 + half * 8 + (i & 7)) * NM + keyb + mrow];
      }
      s0 = __builtin_amdgcn_wmma_f32_16x16x32_bf16(false, aq[0], false, bk0,
                                                   (short)0, s0, false, false);
      s0 = __builtin_amdgcn_wmma_f32_16x16x32_bf16(false, aq[1], false, bk1,
                                                   (short)0, s0, false, false);
      cb0 = mask[(size_t)b * NM + keyb + mrow] ? 0.0f : -1e9f;
    }
    {
      int keyb = kt * 32 + 16;
      v16bf bk0, bk1;
#pragma unroll
      for (int i = 0; i < 16; ++i) {
        bk0[i] = (__bf16)kp[(size_t)((i >> 3) * 16 + half * 8 + (i & 7)) * NM + keyb + mrow];
        bk1[i] = (__bf16)kp[(size_t)(32 + (i >> 3) * 16 + half * 8 + (i & 7)) * NM + keyb + mrow];
      }
      s1 = __builtin_amdgcn_wmma_f32_16x16x32_bf16(false, aq[0], false, bk0,
                                                   (short)0, s1, false, false);
      s1 = __builtin_amdgcn_wmma_f32_16x16x32_bf16(false, aq[1], false, bk1,
                                                   (short)0, s1, false, false);
      cb1 = mask[(size_t)b * NM + keyb + mrow] ? 0.0f : -1e9f;
    }

    float tmax[8];
#pragma unroll
    for (int r = 0; r < 8; ++r) tmax[r] = fmaxf(s0[r] + cb0, s1[r] + cb1);
#pragma unroll
    for (int off = 8; off >= 1; off >>= 1)
#pragma unroll
      for (int r = 0; r < 8; ++r)
        tmax[r] = fmaxf(tmax[r], __shfl_xor(tmax[r], off, 32));

    float psum[8], alph[8];
#pragma unroll
    for (int r = 0; r < 8; ++r) {
      float nm = fmaxf(rmax[r], tmax[r]);
      alph[r] = __expf(rmax[r] - nm);
      rmax[r] = nm;
      float p0 = __expf(s0[r] + cb0 - nm);
      float p1 = __expf(s1[r] + cb1 - nm);
      Pw[wave][r + 8 * half][mrow] = p0;
      Pw[wave][r + 8 * half][16 + mrow] = p1;
      psum[r] = p0 + p1;
    }
#pragma unroll
    for (int off = 8; off >= 1; off >>= 1)
#pragma unroll
      for (int r = 0; r < 8; ++r) psum[r] += __shfl_xor(psum[r], off, 32);
#pragma unroll
    for (int r = 0; r < 8; ++r) rsum[r] = rsum[r] * alph[r] + psum[r];
#pragma unroll
    for (int t = 0; t < 4; ++t)
#pragma unroll
      for (int r = 0; r < 8; ++r) accO[t][r] *= alph[r];

    __syncthreads();

    v16bf pa;
#pragma unroll
    for (int i = 0; i < 16; ++i) {
      int kk = (i >> 3) * 16 + half * 8 + (i & 7);
      pa[i] = (__bf16)Pw[wave][mrow][kk];
    }
#pragma unroll
    for (int t = 0; t < 4; ++t) {
      v16bf bv;
#pragma unroll
      for (int i = 0; i < 16; ++i)
        bv[i] = (__bf16)vp[(size_t)(kt * 32 + (i >> 3) * 16 + half * 8 + (i & 7)) * NHD + t * 16 + mrow];
      accO[t] = __builtin_amdgcn_wmma_f32_16x16x32_bf16(
          false, pa, false, bv, (short)0, accO[t], false, false);
    }
    __syncthreads();
  }

#pragma unroll
  for (int t = 0; t < 4; ++t)
#pragma unroll
    for (int r = 0; r < 8; ++r) {
      int row = r + 8 * half;
      int col = t * 16 + mrow;
      float inv = rsum[r] > 0.0f ? 1.0f / rsum[r] : 0.0f;
      o[((size_t)b * NM + qbase + row) * ND + h * NHD + col] = accO[t][r] * inv;
    }
}

// =====================================================================
// LayerNorm (no affine) + adaLN modulate: out = ln(x)*(1+sc)+sh
// =====================================================================
__global__ void ln_mod_kernel(const float* __restrict__ x,
                              const float* __restrict__ shBase,
                              const float* __restrict__ scBase,
                              int stride, float* __restrict__ out) {
  const int row = blockIdx.x;
  const int b = row / NM;
  const float* xr = x + (size_t)row * ND;
  __shared__ float r1[256], r2[256];
  float s = 0.0f, s2 = 0.0f;
  for (int d = threadIdx.x; d < ND; d += 256) {
    float v = xr[d];
    s += v; s2 += v * v;
  }
  r1[threadIdx.x] = s; r2[threadIdx.x] = s2;
  __syncthreads();
  for (int off = 128; off; off >>= 1) {
    if (threadIdx.x < off) {
      r1[threadIdx.x] += r1[threadIdx.x + off];
      r2[threadIdx.x] += r2[threadIdx.x + off];
    }
    __syncthreads();
  }
  float mu = r1[0] * (1.0f / ND);
  float var = r2[0] * (1.0f / ND) - mu * mu;
  float rs = rsqrtf(var + 1e-6f);
  for (int d = threadIdx.x; d < ND; d += 256) {
    float sc = scBase[(size_t)b * stride + d];
    float sh = shBase[(size_t)b * stride + d];
    out[(size_t)row * ND + d] = (xr[d] - mu) * rs * (1.0f + sc) + sh;
  }
}

// =====================================================================
// qk-norm (LN over head dim) + RoPE. q,v -> (B*H, M, HD); k -> (B*H, HD, M)
// =====================================================================
__global__ void qkrope_kernel(const float* __restrict__ qkv,
                              const float* __restrict__ qnw,
                              const float* __restrict__ qnb,
                              const float* __restrict__ knw,
                              const float* __restrict__ knb,
                              float* __restrict__ qo,
                              float* __restrict__ koT,
                              float* __restrict__ vo) {
  const int idx = blockIdx.x;          // b*H*M + h*M + m
  const int m = idx % NM;
  const int bh = idx / NM;
  const int b = bh / NH, h = bh % NH;
  const int d = threadIdx.x;           // 0..63
  __shared__ float sA[64], sB[64], sv[64];

  const float* base = qkv + ((size_t)b * NM + m) * (3 * ND);
  float qv = base[h * NHD + d];
  float kv = base[ND + h * NHD + d];
  float vv = base[2 * ND + h * NHD + d];

  int j = d & 31;
  float inv = __expf(-logf(10000.0f) * (float)j / 32.0f);
  float ang = (float)m * inv;
  float cs = cosf(ang), sn = sinf(ang);
  size_t obase = (size_t)bh * NM * NHD + (size_t)m * NHD;

  sA[d] = qv; sB[d] = qv * qv;
  __syncthreads();
  for (int off = 32; off; off >>= 1) {
    if (d < off) { sA[d] += sA[d + off]; sB[d] += sB[d + off]; }
    __syncthreads();
  }
  {
    float mu = sA[0] * (1.0f / 64.0f);
    float var = sB[0] * (1.0f / 64.0f) - mu * mu;
    float rs = rsqrtf(var + 1e-6f);
    sv[d] = (qv - mu) * rs * qnw[d] + qnb[d];
  }
  __syncthreads();
  {
    float outv = (d < 32) ? (sv[d] * cs - sv[d + 32] * sn)
                          : (sv[d - 32] * sn + sv[d] * cs);
    qo[obase + d] = outv;
  }
  __syncthreads();

  sA[d] = kv; sB[d] = kv * kv;
  __syncthreads();
  for (int off = 32; off; off >>= 1) {
    if (d < off) { sA[d] += sA[d + off]; sB[d] += sB[d + off]; }
    __syncthreads();
  }
  {
    float mu = sA[0] * (1.0f / 64.0f);
    float var = sB[0] * (1.0f / 64.0f) - mu * mu;
    float rs = rsqrtf(var + 1e-6f);
    sv[d] = (kv - mu) * rs * knw[d] + knb[d];
  }
  __syncthreads();
  {
    float outv = (d < 32) ? (sv[d] * cs - sv[d + 32] * sn)
                          : (sv[d - 32] * sn + sv[d] * cs);
    koT[(size_t)bh * NHD * NM + (size_t)d * NM + m] = outv;  // transposed
  }
  vo[obase + d] = vv;
}

// gated residual: h += g[b,d] * x
__global__ void resid_kernel(float* __restrict__ hbuf,
                             const float* __restrict__ x,
                             const float* __restrict__ gBase, int stride) {
  size_t idx = (size_t)blockIdx.x * blockDim.x + threadIdx.x;
  if (idx >= (size_t)NB * NM * ND) return;
  int b = (int)(idx / ((size_t)NM * ND));
  int d = (int)(idx % ND);
  hbuf[idx] += gBase[(size_t)b * stride + d] * x[idx];
}

// token embedding modulate: h = t1*(1+scale_a)+shift_a
__global__ void embed_mod_kernel(const float* __restrict__ t1,
                                 const int* __restrict__ at,
                                 const float* __restrict__ atom_emb,
                                 float* __restrict__ h) {
  size_t idx = (size_t)blockIdx.x * blockDim.x + threadIdx.x;
  if (idx >= (size_t)NB * NM * ND) return;
  size_t bm = idx / ND;
  int d = (int)(idx % ND);
  int a = at[bm];
  float sh = atom_emb[(size_t)a * 2 * ND + d];
  float sc = atom_emb[(size_t)a * 2 * ND + ND + d];
  h[idx] = t1[idx] * (1.0f + sc) + sh;
}

// naive GEMM for small / skinny shapes; act: 0 none, 2 silu
__global__ void naive_gemm(const float* __restrict__ A,
                           const float* __restrict__ W,
                           const float* __restrict__ bias,
                           float* __restrict__ C,
                           int M, int N, int K, int act) {
  int total = M * N;
  for (int idx = blockIdx.x * blockDim.x + threadIdx.x; idx < total;
       idx += gridDim.x * blockDim.x) {
    int row = idx / N, n = idx % N;
    float acc = bias ? bias[n] : 0.0f;
    for (int kk = 0; kk < K; ++kk)
      acc += A[(size_t)row * K + kk] * W[(size_t)kk * N + n];
    if (act == 2) acc = siluf(acc);
    C[idx] = acc;
  }
}

// timestep embedding -> hidden = silu(tf @ wt1 + bt1); out (B,5,D)
__global__ void tehid_kernel(const float* __restrict__ t5,
                             const float* __restrict__ wt1,
                             const float* __restrict__ bt1,
                             float* __restrict__ hid) {
  int idx = blockIdx.x * blockDim.x + threadIdx.x;
  if (idx >= NB * 5 * ND) return;
  int d = idx % ND;
  int bs = idx / ND;
  float t = t5[bs];
  float acc = bt1[d];
  for (int f = 0; f < NFREQ; ++f) {
    int j = f & 127;
    float freq = __expf(-logf(10000.0f) * (float)j / 128.0f);
    float a = t * freq;
    float val = (f < 128) ? cosf(a) : sinf(a);
    acc += val * wt1[(size_t)f * ND + d];
  }
  hid[idx] = siluf(acc);
}

// c = sum_s(hid@wt2)+5*bt2 + ds_emb + sg_emb ; store silu(c)
__global__ void cond_kernel(const float* __restrict__ hid,
                            const float* __restrict__ wt2,
                            const float* __restrict__ bt2,
                            const float* __restrict__ ds_emb,
                            const float* __restrict__ sg_emb,
                            const int* __restrict__ dsi,
                            const int* __restrict__ sg,
                            float* __restrict__ scOut) {
  int idx = blockIdx.x * blockDim.x + threadIdx.x;
  if (idx >= NB * ND) return;
  int b = idx / ND, d = idx % ND;
  float acc = 5.0f * bt2[d];
  acc += ds_emb[(size_t)dsi[b] * ND + d];
  acc += sg_emb[(size_t)sg[b] * ND + d];
  for (int s = 0; s < 5; ++s) {
    const float* hr = hid + ((size_t)b * 5 + s) * ND;
    float a = 0.0f;
    for (int jj = 0; jj < ND; ++jj) a += hr[jj] * wt2[(size_t)jj * ND + d];
    acc += a;
  }
  scOut[idx] = siluf(acc);
}

// masked mean pool over M: pooled (B, D)
__global__ void pool_kernel(const float* __restrict__ hn,
                            const unsigned char* __restrict__ mask,
                            float* __restrict__ pooled) {
  int idx = blockIdx.x * blockDim.x + threadIdx.x;
  if (idx >= NB * ND) return;
  int b = idx / ND, d = idx % ND;
  float s = 0.0f, cnt = 0.0f;
  for (int m = 0; m < NM; ++m) {
    if (mask[(size_t)b * NM + m]) {
      s += hn[((size_t)b * NM + m) * ND + d];
      cnt += 1.0f;
    }
  }
  pooled[idx] = s / fmaxf(cnt, 1.0f);
}

// =====================================================================
extern "C" void kernel_launch(void* const* d_in, const int* in_sizes, int n_in,
                              void* d_out, int out_size, void* d_ws,
                              size_t ws_size, hipStream_t stream) {
  (void)in_sizes; (void)n_in; (void)out_size; (void)ws_size;

  const int*   at    = (const int*)d_in[0];
  const float* xf    = (const float*)d_in[1];
  const float* t5    = (const float*)d_in[2];
  const int*   dsi   = (const int*)d_in[3];
  const int*   sg    = (const int*)d_in[4];
  const unsigned char* mask = (const unsigned char*)d_in[5];
  const float* ada_w = (const float*)d_in[6];
  const float* ada_b = (const float*)d_in[7];
  const float* wqkv  = (const float*)d_in[8];
  const float* qn_w  = (const float*)d_in[9];
  const float* qn_b  = (const float*)d_in[10];
  const float* kn_w  = (const float*)d_in[11];
  const float* kn_b  = (const float*)d_in[12];
  const float* wproj = (const float*)d_in[13];
  const float* w1    = (const float*)d_in[14];
  const float* b1    = (const float*)d_in[15];
  const float* w2    = (const float*)d_in[16];
  const float* b2    = (const float*)d_in[17];
  const float* atom_emb = (const float*)d_in[18];
  const float* wx    = (const float*)d_in[19];
  const float* bx    = (const float*)d_in[20];
  const float* wt1   = (const float*)d_in[21];
  const float* bt1   = (const float*)d_in[22];
  const float* wt2   = (const float*)d_in[23];
  const float* bt2   = (const float*)d_in[24];
  const float* ds_emb = (const float*)d_in[25];
  const float* sg_emb = (const float*)d_in[26];
  const float* fada_w = (const float*)d_in[27];
  const float* fada_b = (const float*)d_in[28];
  const float* w_at  = (const float*)d_in[29];
  const float* b_at  = (const float*)d_in[30];
  const float* w_pos = (const float*)d_in[31];
  const float* w_fc  = (const float*)d_in[32];
  const float* w_len = (const float*)d_in[33];
  const float* b_len = (const float*)d_in[34];
  const float* w_ang = (const float*)d_in[35];
  const float* b_ang = (const float*)d_in[36];

  float* ws = (float*)d_ws;
  size_t o = 0;
  float* WS_HID  = ws + o; o += (size_t)NB * 5 * ND;
  float* WS_SC   = ws + o; o += (size_t)NB * ND;
  float* WS_MOD  = ws + o; o += (size_t)NB * 6 * ND;
  float* WS_POOL = ws + o; o += (size_t)NB * ND;
  float* WS_FMOD = ws + o; o += (size_t)NB * 2 * ND;
  float* WS_H    = ws + o; o += (size_t)NB * NM * ND;
  float* WS_T1   = ws + o; o += (size_t)NB * NM * ND;
  float* WS_Q    = ws + o; o += (size_t)NB * NM * ND;
  float* WS_K    = ws + o; o += (size_t)NB * NM * ND;
  float* WS_V    = ws + o; o += (size_t)NB * NM * ND;
  float* WS_O    = ws + o; o += (size_t)NB * NM * ND;
  float* WS_BIG  = ws + o; o += (size_t)NB * NM * NMLP;

  float* out = (float*)d_out;
  const size_t OFF_POS  = (size_t)NB * NM * NV;
  const size_t OFF_FRAC = OFF_POS + (size_t)NB * NM * 3;
  const size_t OFF_LEN  = OFF_FRAC + (size_t)NB * NM * 3;
  const size_t OFF_ANG  = OFF_LEN + (size_t)NB * 3;

  const int TOK = NB * NM;  // 8192

  auto gemm = [&](const float* A, const float* W, const float* bias, float* C,
                  int M, int N, int K, int act) {
    dim3 g(N / 128, M / 128);
    gemm_wmma_fast<<<g, dim3(256), 0, stream>>>(A, W, bias, C, M, N, K, act);
  };
  auto gemmT = [&](const float* A, const float* W, const float* bias, float* C,
                   int M, int N, int K, int act) {
    dim3 g(N / 128, M / 128);
    gemm_wmma_tdm<<<g, dim3(256), 0, stream>>>(A, W, bias, C, M, N, K, act);
  };

  // ---- conditioning ----
  tehid_kernel<<<(NB * 5 * ND + 255) / 256, 256, 0, stream>>>(t5, wt1, bt1, WS_HID);
  cond_kernel<<<(NB * ND + 255) / 256, 256, 0, stream>>>(
      WS_HID, wt2, bt2, ds_emb, sg_emb, dsi, sg, WS_SC);

  // ---- token embedding ----
  gemm(xf, wx, bx, WS_T1, TOK, ND, NDX, 0);
  embed_mod_kernel<<<(TOK * ND + 255) / 256, 256, 0, stream>>>(
      WS_T1, at, atom_emb, WS_H);

  // ---- transformer blocks ----
  for (int l = 0; l < NL; ++l) {
    const float* aw = ada_w + (size_t)l * ND * 6 * ND;
    const float* ab = ada_b + (size_t)l * 6 * ND;
    naive_gemm<<<256, 256, 0, stream>>>(WS_SC, aw, ab, WS_MOD, NB, 6 * ND, ND, 0);

    ln_mod_kernel<<<TOK, 256, 0, stream>>>(WS_H, WS_MOD, WS_MOD + ND, 6 * ND, WS_T1);
    gemm(WS_T1, wqkv + (size_t)l * ND * 3 * ND, nullptr, WS_BIG, TOK, 3 * ND, ND, 0);

    qkrope_kernel<<<NB * NH * NM, 64, 0, stream>>>(
        WS_BIG, qn_w + (size_t)l * NHD, qn_b + (size_t)l * NHD,
        kn_w + (size_t)l * NHD, kn_b + (size_t)l * NHD, WS_Q, WS_K, WS_V);

    attn_wmma<<<NB * NH * (NM / 64), 128, 0, stream>>>(WS_Q, WS_K, WS_V, mask, WS_O);

    gemm(WS_O, wproj + (size_t)l * ND * ND, nullptr, WS_T1, TOK, ND, ND, 0);
    resid_kernel<<<(TOK * ND + 255) / 256, 256, 0, stream>>>(
        WS_H, WS_T1, WS_MOD + 2 * ND, 6 * ND);

    ln_mod_kernel<<<TOK, 256, 0, stream>>>(WS_H, WS_MOD + 3 * ND, WS_MOD + 4 * ND,
                                           6 * ND, WS_T1);
    gemmT(WS_T1, w1 + (size_t)l * ND * NMLP, b1 + (size_t)l * NMLP, WS_BIG,
          TOK, NMLP, ND, 1 /*gelu*/);
    gemmT(WS_BIG, w2 + (size_t)l * NMLP * ND, b2 + (size_t)l * ND, WS_T1,
          TOK, ND, NMLP, 0);
    resid_kernel<<<(TOK * ND + 255) / 256, 256, 0, stream>>>(
        WS_H, WS_T1, WS_MOD + 5 * ND, 6 * ND);
  }

  // ---- final adaLN + heads ----
  naive_gemm<<<96, 256, 0, stream>>>(WS_SC, fada_w, fada_b, WS_FMOD, NB, 2 * ND, ND, 0);
  ln_mod_kernel<<<TOK, 256, 0, stream>>>(WS_H, WS_FMOD, WS_FMOD + ND, 2 * ND, WS_T1);

  {
    dim3 g((NV + 127) / 128, TOK / 128);
    gemm_wmma_nck<<<g, dim3(256), 0, stream>>>(WS_T1, w_at, b_at, out,
                                               TOK, NV, ND, 0);
  }
  naive_gemm<<<96, 256, 0, stream>>>(WS_T1, w_pos, nullptr, out + OFF_POS, TOK, 3, ND, 0);
  naive_gemm<<<96, 256, 0, stream>>>(WS_T1, w_fc, nullptr, out + OFF_FRAC, TOK, 3, ND, 0);

  pool_kernel<<<(NB * ND + 255) / 256, 256, 0, stream>>>(WS_T1, mask, WS_POOL);
  naive_gemm<<<1, 64, 0, stream>>>(WS_POOL, w_len, b_len, out + OFF_LEN, NB, 3, ND, 0);
  naive_gemm<<<1, 64, 0, stream>>>(WS_POOL, w_ang, b_ang, out + OFF_ANG, NB, 3, ND, 0);
}
